// SparseExpertV3_88665304858729
// MI455X (gfx1250) — matrix operand order repaired and checked by reading, not compiled, not run on hardware
//
#include <hip/hip_runtime.h>
#include <hip/hip_bf16.h>
#include <math.h>

// Problem constants: x:(4,2048,2048) -> N=8192 rows of D=2048
#define D_DIM   2048
#define M_EXP   64
#define B_DIM   64
#define K_TOP   8
#define N_ROWS  8192
#define J_COLS  4096                       // M_EXP * B_DIM
#define N_ENT   (N_ROWS * K_TOP)           // 65536 (row,expert) entries
#define MAXTILE (N_ENT / 16 + M_EXP)       // 4160 upper bound on 16-entry tiles
#define ALPHA_C 1.0f
#define EPS_C   1e-8f

typedef __attribute__((ext_vector_type(16))) __bf16 v16bf;
typedef __attribute__((ext_vector_type(8)))  __bf16 v8bf;
typedef __attribute__((ext_vector_type(8)))  float  v8f;

// ---------------- workspace layout (bytes), total ~347 MB -----------------
static const size_t OFF_XN     = 0;                                          // f32  N*D      64 MB
static const size_t OFF_XNB    = OFF_XN     + (size_t)N_ROWS*D_DIM*4;        // bf16 N*D      32 MB
static const size_t OFF_VTB    = OFF_XNB    + (size_t)N_ROWS*D_DIM*2;        // bf16 (j,d)    16 MB
static const size_t OFF_VNB    = OFF_VTB    + (size_t)J_COLS*D_DIM*2;        // bf16 (d,j)    16 MB
static const size_t OFF_UNB    = OFF_VNB    + (size_t)D_DIM*J_COLS*2;        // bf16 (m,d,b)  16 MB
static const size_t OFF_UTB    = OFF_UNB    + (size_t)M_EXP*D_DIM*B_DIM*2;   // bf16 (j,d)    16 MB
static const size_t OFF_ENERGY = OFF_UTB    + (size_t)J_COLS*D_DIM*2;        // f32  N*M       2 MB
static const size_t OFF_ESUM   = OFF_ENERGY + (size_t)N_ROWS*M_EXP*4;        // f32  M
static const size_t OFF_TOPK   = OFF_ESUM   + 256;                           // i32  N*K     256 KB
static const size_t OFF_HSD    = OFF_TOPK   + (size_t)N_ENT*4;               // f32  N*K*B    16 MB
static const size_t OFF_HSB    = OFF_HSD    + (size_t)N_ENT*B_DIM*4;         // bf16 entry*B   8 MB
static const size_t OFF_WRITES = OFF_HSB    + (size_t)N_ENT*B_DIM*2;         // f32  N*D      64 MB
static const size_t OFF_XHAT   = OFF_WRITES + (size_t)N_ROWS*D_DIM*4;        // f32  N*D      64 MB
static const size_t OFF_WRB    = OFF_XHAT   + (size_t)N_ROWS*D_DIM*4;        // bf16 N*D      32 MB
static const size_t OFF_ROWS   = OFF_WRB    + (size_t)N_ROWS*D_DIM*2;        // i32  N*K     256 KB
static const size_t OFF_SLOT   = OFF_ROWS   + (size_t)N_ENT*4;               // i32  N*K     256 KB
static const size_t OFF_COUNTS = OFF_SLOT   + (size_t)N_ENT*4;               // i32  M
static const size_t OFF_OFFS   = OFF_COUNTS + 256;                           // i32  M+1
static const size_t OFF_CURS   = OFF_OFFS   + 512;                           // i32  M
static const size_t OFF_TILEM  = OFF_CURS   + 256;                           // i32  MAXTILE
static const size_t OFF_TILES  = OFF_TILEM  + (size_t)MAXTILE*4;             // i32  MAXTILE
static const size_t OFF_TCNT   = OFF_TILES  + (size_t)MAXTILE*4;             // i32  1
static const size_t OFF_ACCUM  = OFF_TCNT   + 256;                           // f32  [0]=resid [1]=writer

// ---------------- helpers -----------------
__device__ inline float blockReduceSum256(float v, float* s) {
    int t = threadIdx.x;
    s[t] = v; __syncthreads();
    for (int off = 128; off > 0; off >>= 1) {
        if (t < off) s[t] += s[t + off];
        __syncthreads();
    }
    float r = s[0]; __syncthreads(); return r;
}
__device__ inline float blockReduceSum128(float v, float* s) {
    int t = threadIdx.x;
    s[t] = v; __syncthreads();
    for (int off = 64; off > 0; off >>= 1) {
        if (t < off) s[t] += s[t + off];
        __syncthreads();
    }
    float r = s[0]; __syncthreads(); return r;
}

// ---------------- normalization kernels -----------------
__global__ void __launch_bounds__(256)
k_norm_x(const float* __restrict__ x, float* __restrict__ xn,
         __hip_bfloat16* __restrict__ xnb) {
    __shared__ float red[256];
    int n = blockIdx.x, t = threadIdx.x;
    const float* row = x + (size_t)n * D_DIM;
    float s = 0.f;
    for (int d = t; d < D_DIM; d += 256) { float v = row[d]; s += v * v; }
    float tot = blockReduceSum256(s, red);
    float inv = 1.f / fmaxf(sqrtf(tot), EPS_C);
    for (int d = t; d < D_DIM; d += 256) {
        float v = row[d] * inv;
        xn [(size_t)n * D_DIM + d] = v;
        xnb[(size_t)n * D_DIM + d] = __float2bfloat16(v);
    }
}

// V:(D,M,B) normalized over D per column j=(m,b); emit bf16 in (j,d) and (d,j) layouts
__global__ void __launch_bounds__(256)
k_norm_v(const float* __restrict__ V, __hip_bfloat16* __restrict__ vtb,
         __hip_bfloat16* __restrict__ vnb) {
    __shared__ float red[256];
    int j = blockIdx.x, t = threadIdx.x;
    float s = 0.f;
    for (int d = t; d < D_DIM; d += 256) { float v = V[(size_t)d * J_COLS + j]; s += v * v; }
    float tot = blockReduceSum256(s, red);
    float inv = 1.f / fmaxf(sqrtf(tot), 1e-12f);
    for (int d = t; d < D_DIM; d += 256) {
        float v = V[(size_t)d * J_COLS + j] * inv;
        __hip_bfloat16 bv = __float2bfloat16(v);
        vtb[(size_t)j * D_DIM + d]  = bv;   // (j,d): K=d contiguous
        vnb[(size_t)d * J_COLS + j] = bv;   // (d,j): K=b contiguous
    }
}

// U:(M,D,B) normalized over D (stride B); emit bf16 in (m,d,b) and transposed (j,d) layouts
__global__ void __launch_bounds__(256)
k_norm_u(const float* __restrict__ U, __hip_bfloat16* __restrict__ unb,
         __hip_bfloat16* __restrict__ utb) {
    __shared__ float red[256];
    int mb = blockIdx.x, t = threadIdx.x;
    int m = mb >> 6, b = mb & 63;
    const float* base = U + (size_t)m * D_DIM * B_DIM + b;
    float s = 0.f;
    for (int d = t; d < D_DIM; d += 256) { float v = base[(size_t)d * B_DIM]; s += v * v; }
    float tot = blockReduceSum256(s, red);
    float inv = 1.f / fmaxf(sqrtf(tot), 1e-12f);
    for (int d = t; d < D_DIM; d += 256) {
        float v = base[(size_t)d * B_DIM] * inv;
        __hip_bfloat16 bv = __float2bfloat16(v);
        unb[((size_t)m * D_DIM + d) * B_DIM + b] = bv;  // (m,d,b): K=b contiguous
        utb[(size_t)mb * D_DIM + d]              = bv;  // (j,d):   K=d contiguous
    }
}

// ---------------- dense WMMA GEMM -> per-expert energy -----------------
// wave = one 16x16 tile of h = xn(8192x2048) * Vn(2048x4096); block = 16 rows x 64 cols
__global__ void __launch_bounds__(128)
k_gemm_energy(const __bf16* __restrict__ xnb, const __bf16* __restrict__ vtb,
              float* __restrict__ energy) {
    int lane = threadIdx.x & 31;
    int wave = threadIdx.x >> 5;
    int n0 = blockIdx.y * 16;
    int j0 = blockIdx.x * 64 + wave * 16;
    int kbA = (lane < 16) ? 0 : 8;
    int kbB = (lane < 16) ? 0 : 8;
    const __bf16* arow = xnb + (size_t)(n0 + (lane & 15)) * D_DIM;
    const __bf16* brow = vtb + (size_t)(j0 + (lane & 15)) * D_DIM;
    v8f c = {};
    for (int k0 = 0; k0 < D_DIM; k0 += 32) {
        union { v16bf v; v8bf h[2]; } a, b;
        a.h[0] = *(const v8bf*)(arow + k0 + kbA);
        a.h[1] = *(const v8bf*)(arow + k0 + kbA + 16);
        b.h[0] = *(const v8bf*)(brow + k0 + kbB);
        b.h[1] = *(const v8bf*)(brow + k0 + kbB + 16);
        __builtin_prefetch(arow + k0 + 256, 0, 0);
        __builtin_prefetch(brow + k0 + 256, 0, 0);
        c = __builtin_amdgcn_wmma_f32_16x16x32_bf16(false, a.v, false, b.v,
                                                    (short)0, c, false, false);
    }
    int e = j0 >> 6;
    int rbase = (lane < 16) ? 0 : 8;
#pragma unroll
    for (int r = 0; r < 8; ++r) {
        float v = c[r];
        atomicAdd(&energy[(size_t)(n0 + rbase + r) * M_EXP + e], v * v);
    }
}

// ---------------- top-K + expert energy sums -----------------
__global__ void k_topk(const float* __restrict__ energy, int* __restrict__ topk) {
    int n = blockIdx.x * blockDim.x + threadIdx.x;
    if (n >= N_ROWS) return;
    const float* e = energy + (size_t)n * M_EXP;
    unsigned long long chosen = 0ull;
    for (int k = 0; k < K_TOP; ++k) {
        float best = -1e30f; int bi = 0;
        for (int m = 0; m < M_EXP; ++m) {
            if ((chosen >> m) & 1ull) continue;
            float v = e[m];
            if (v > best) { best = v; bi = m; }
        }
        chosen |= (1ull << bi);
        topk[(size_t)n * K_TOP + k] = bi;
    }
}

__global__ void __launch_bounds__(256)
k_energy_sum(const float* __restrict__ energy, float* __restrict__ esum) {
    __shared__ float red[256];
    int m = blockIdx.x, t = threadIdx.x;
    float s = 0.f;
    for (int n = t; n < N_ROWS; n += 256) s += energy[(size_t)n * M_EXP + m];
    float tot = blockReduceSum256(s, red);
    if (t == 0) esum[m] = tot;
}

// ---------------- CSR build: (row,expert) entries grouped by expert -----------------
__global__ void k_count(const int* __restrict__ topk, int* __restrict__ counts) {
    int i = blockIdx.x * 256 + threadIdx.x;
    if (i < N_ENT) atomicAdd(&counts[topk[i]], 1);
}

__global__ void k_scan(const int* __restrict__ counts, int* __restrict__ offs,
                       int* __restrict__ tileM, int* __restrict__ tileS,
                       int* __restrict__ tcnt) {
    if (threadIdx.x != 0 || blockIdx.x != 0) return;
    int off = 0;
    for (int m = 0; m < M_EXP; ++m) { offs[m] = off; off += counts[m]; }
    offs[M_EXP] = off;
    int nt = 0;
    for (int m = 0; m < M_EXP; ++m) {
        int c = counts[m];
        for (int s = 0; s < c; s += 16) {
            tileM[nt] = m;
            tileS[nt] = offs[m] + s;
            ++nt;
        }
    }
    tcnt[0] = nt;
}

__global__ void k_fill(const int* __restrict__ topk, const int* __restrict__ offs,
                       int* __restrict__ cursor, int* __restrict__ rows,
                       int* __restrict__ slot) {
    int i = blockIdx.x * 256 + threadIdx.x;
    if (i >= N_ENT) return;
    int m = topk[i];
    int pos = atomicAdd(&cursor[m], 1);
    int e = offs[m] + pos;
    rows[e] = i / K_TOP;
    slot[e] = i;                 // n*K + k  -> h_sparse slot
}

// ---------------- expert-blocked WMMA: h_sparse = gathered xn rows x Vn_m --------
// block: one 16-entry tile, 4 waves cover b=0..63. K = D = 2048.
__global__ void __launch_bounds__(128)
k_hsparse(const __bf16* __restrict__ xnb, const __bf16* __restrict__ vtb,
          const int* __restrict__ tileM, const int* __restrict__ tileS,
          const int* __restrict__ tcnt, const int* __restrict__ offs,
          const int* __restrict__ rows, const int* __restrict__ slot,
          float* __restrict__ hsd, __hip_bfloat16* __restrict__ hsb) {
    int tile = blockIdx.x;
    if (tile >= tcnt[0]) return;                   // uniform -> EXEC stays all-ones
    int m = tileM[tile], estart = tileS[tile], eend = offs[m + 1];
    int lane = threadIdx.x & 31;
    int wave = threadIdx.x >> 5;
    int b0 = wave * 16;
    int l15 = lane & 15;
    int ec = min(estart + l15, eend - 1);          // clamp gather (stores masked)
    int kbA = (lane < 16) ? 0 : 8;
    int kbB = (lane < 16) ? 0 : 8;
    const __bf16* arow = xnb + (size_t)rows[ec] * D_DIM;
    const __bf16* brow = vtb + (size_t)(m * B_DIM + b0 + l15) * D_DIM;
    v8f c = {};
    for (int k0 = 0; k0 < D_DIM; k0 += 32) {
        union { v16bf v; v8bf h[2]; } a, b;
        a.h[0] = *(const v8bf*)(arow + k0 + kbA);
        a.h[1] = *(const v8bf*)(arow + k0 + kbA + 16);
        b.h[0] = *(const v8bf*)(brow + k0 + kbB);
        b.h[1] = *(const v8bf*)(brow + k0 + kbB + 16);
        __builtin_prefetch(brow + k0 + 256, 0, 0);
        c = __builtin_amdgcn_wmma_f32_16x16x32_bf16(false, a.v, false, b.v,
                                                    (short)0, c, false, false);
    }
    int rbase = (lane < 16) ? 0 : 8;
#pragma unroll
    for (int r = 0; r < 8; ++r) {
        int ee = estart + rbase + r;
        if (ee < eend) {
            float v = c[r];
            hsd[(size_t)slot[ee] * B_DIM + b0 + l15] = v;
            hsb[(size_t)ee * B_DIM + b0 + l15] = __float2bfloat16(v);
        }
    }
}

// ---------------- expert-blocked WMMA: x_hat & writes scatter -----------------
// C = h_m(16 x 64) x {Vn_m, Un_m}(64 x 2048); block covers 64 d-cols; accumulate with atomics.
__global__ void __launch_bounds__(128)
k_scatter(const __bf16* __restrict__ hsb, const __bf16* __restrict__ vnb,
          const __bf16* __restrict__ unb,
          const int* __restrict__ tileM, const int* __restrict__ tileS,
          const int* __restrict__ tcnt, const int* __restrict__ offs,
          const int* __restrict__ rows,
          float* __restrict__ xhat, float* __restrict__ writes) {
    int tile = blockIdx.x;
    if (tile >= tcnt[0]) return;
    int m = tileM[tile], estart = tileS[tile], eend = offs[m + 1];
    int lane = threadIdx.x & 31;
    int wave = threadIdx.x >> 5;
    int d0 = blockIdx.y * 64 + wave * 16;
    int l15 = lane & 15;
    int ec = min(estart + l15, eend - 1);
    int kbA = (lane < 16) ? 0 : 8;
    int kbB = (lane < 16) ? 0 : 8;
    const __bf16* arow  = hsb + (size_t)ec * B_DIM;
    const __bf16* browV = vnb + (size_t)(d0 + l15) * J_COLS + m * B_DIM;   // K=b stride 1
    const __bf16* browU = unb + ((size_t)m * D_DIM + (d0 + l15)) * B_DIM;  // K=b stride 1
    v8f cx = {}, cw = {};
#pragma unroll
    for (int k0 = 0; k0 < B_DIM; k0 += 32) {
        union { v16bf v; v8bf h[2]; } a, bv, bu;
        a.h[0]  = *(const v8bf*)(arow + k0 + kbA);
        a.h[1]  = *(const v8bf*)(arow + k0 + kbA + 16);
        bv.h[0] = *(const v8bf*)(browV + k0 + kbB);
        bv.h[1] = *(const v8bf*)(browV + k0 + kbB + 16);
        bu.h[0] = *(const v8bf*)(browU + k0 + kbB);
        bu.h[1] = *(const v8bf*)(browU + k0 + kbB + 16);
        cx = __builtin_amdgcn_wmma_f32_16x16x32_bf16(false, a.v, false, bv.v,
                                                     (short)0, cx, false, false);
        cw = __builtin_amdgcn_wmma_f32_16x16x32_bf16(false, a.v, false, bu.v,
                                                     (short)0, cw, false, false);
    }
    int rbase = (lane < 16) ? 0 : 8;
#pragma unroll
    for (int r = 0; r < 8; ++r) {
        int ee = estart + rbase + r;
        if (ee < eend) {
            size_t idx = (size_t)rows[ee] * D_DIM + d0 + l15;
            atomicAdd(&xhat[idx],   cx[r]);
            atomicAdd(&writes[idx], cw[r]);
        }
    }
}

// ---------------- per-row epilogue: residual energy, x_out, bf16 writes --------
__global__ void __launch_bounds__(256)
k_postrow(const float* __restrict__ xn, const float* __restrict__ xhat,
          const float* __restrict__ writes, __hip_bfloat16* __restrict__ wrb,
          float* __restrict__ xout, float* __restrict__ accum) {
    __shared__ float red[256];
    int n = blockIdx.x, t = threadIdx.x;
    float rss = 0.f, yss = 0.f;
    float yv[D_DIM / 256];
#pragma unroll
    for (int i = 0; i < D_DIM / 256; ++i) {
        size_t idx = (size_t)n * D_DIM + t + i * 256;
        float w = writes[idx];
        wrb[idx] = __float2bfloat16(w);
        float r = xn[idx] - xhat[idx]; rss += r * r;
        float y = xn[idx] + ALPHA_C * w; yv[i] = y; yss += y * y;
    }
    float ytot = blockReduceSum256(yss, red);
    float rtot = blockReduceSum256(rss, red);
    float inv = 1.f / fmaxf(sqrtf(ytot), EPS_C);
#pragma unroll
    for (int i = 0; i < D_DIM / 256; ++i)
        xout[(size_t)n * D_DIM + t + i * 256] = yv[i] * inv;
    if (t == 0) atomicAdd(&accum[0], rtot);
}

// ---------------- expert-blocked WMMA: h_recon + writer loss -----------------
// C = gathered writes rows (16 x 2048) x Un_m^T (2048 x 64)
__global__ void __launch_bounds__(128)
k_hrecon(const __bf16* __restrict__ wrb, const __bf16* __restrict__ utb,
         const int* __restrict__ tileM, const int* __restrict__ tileS,
         const int* __restrict__ tcnt, const int* __restrict__ offs,
         const int* __restrict__ rows, const int* __restrict__ slot,
         const float* __restrict__ hsd, float* __restrict__ accum) {
    __shared__ float red[128];
    int tile = blockIdx.x;
    if (tile >= tcnt[0]) return;
    int m = tileM[tile], estart = tileS[tile], eend = offs[m + 1];
    int lane = threadIdx.x & 31;
    int wave = threadIdx.x >> 5;
    int b0 = wave * 16;
    int l15 = lane & 15;
    int ec = min(estart + l15, eend - 1);
    int kbA = (lane < 16) ? 0 : 8;
    int kbB = (lane < 16) ? 0 : 8;
    const __bf16* arow = wrb + (size_t)rows[ec] * D_DIM;
    const __bf16* brow = utb + (size_t)(m * B_DIM + b0 + l15) * D_DIM;
    v8f c = {};
    for (int k0 = 0; k0 < D_DIM; k0 += 32) {
        union { v16bf v; v8bf h[2]; } a, b;
        a.h[0] = *(const v8bf*)(arow + k0 + kbA);
        a.h[1] = *(const v8bf*)(arow + k0 + kbA + 16);
        b.h[0] = *(const v8bf*)(brow + k0 + kbB);
        b.h[1] = *(const v8bf*)(brow + k0 + kbB + 16);
        __builtin_prefetch(brow + k0 + 256, 0, 0);
        c = __builtin_amdgcn_wmma_f32_16x16x32_bf16(false, a.v, false, b.v,
                                                    (short)0, c, false, false);
    }
    int rbase = (lane < 16) ? 0 : 8;
    float loc = 0.f;
#pragma unroll
    for (int r = 0; r < 8; ++r) {
        int ee = estart + rbase + r;
        if (ee < eend) {
            float diff = c[r] - hsd[(size_t)slot[ee] * B_DIM + b0 + l15];
            loc += diff * diff;
        }
    }
    float tot = blockReduceSum128(loc, red);
    if (threadIdx.x == 0) atomicAdd(&accum[1], tot);
}

// ---------------- finalize aux loss -----------------
__global__ void k_finalize(const float* __restrict__ esum,
                           const float* __restrict__ accum,
                           float* __restrict__ out_aux) {
    __shared__ float red[64];
    int m = threadIdx.x;
    float avg = esum[m] / (float)N_ROWS;
    red[m] = avg; __syncthreads();
    for (int off = 32; off > 0; off >>= 1) {
        if (m < off) red[m] += red[m + off];
        __syncthreads();
    }
    float denom = fmaxf(red[0], EPS_C);
    __syncthreads();
    float p = fmaxf(avg / denom, EPS_C);
    red[m] = p * logf(p); __syncthreads();
    for (int off = 32; off > 0; off >>= 1) {
        if (m < off) red[m] += red[m + off];
        __syncthreads();
    }
    if (m == 0) {
        float ent   = -red[0] / logf((float)M_EXP);
        float uncap = accum[0] / (float)N_ROWS;
        float wloss = accum[1] / (float)((size_t)N_ROWS * K_TOP * B_DIM);
        out_aux[0] = uncap + 0.5f * (1.f - ent) + wloss;
    }
}

// ---------------- host launch -----------------
extern "C" void kernel_launch(void* const* d_in, const int* in_sizes, int n_in,
                              void* d_out, int out_size, void* d_ws, size_t ws_size,
                              hipStream_t stream) {
    const float* x = (const float*)d_in[0];
    const float* V = (const float*)d_in[1];
    const float* U = (const float*)d_in[2];
    float* out = (float*)d_out;

    char* ws = (char*)d_ws;
    float*          xn     = (float*)         (ws + OFF_XN);
    __hip_bfloat16* xnb    = (__hip_bfloat16*)(ws + OFF_XNB);
    __hip_bfloat16* vtb    = (__hip_bfloat16*)(ws + OFF_VTB);
    __hip_bfloat16* vnb    = (__hip_bfloat16*)(ws + OFF_VNB);
    __hip_bfloat16* unb    = (__hip_bfloat16*)(ws + OFF_UNB);
    __hip_bfloat16* utb    = (__hip_bfloat16*)(ws + OFF_UTB);
    float*          energy = (float*)         (ws + OFF_ENERGY);
    float*          esum   = (float*)         (ws + OFF_ESUM);
    int*            topk   = (int*)           (ws + OFF_TOPK);
    float*          hsd    = (float*)         (ws + OFF_HSD);
    __hip_bfloat16* hsb    = (__hip_bfloat16*)(ws + OFF_HSB);
    float*          writes = (float*)         (ws + OFF_WRITES);
    float*          xhat   = (float*)         (ws + OFF_XHAT);
    __hip_bfloat16* wrb    = (__hip_bfloat16*)(ws + OFF_WRB);
    int*            rows   = (int*)           (ws + OFF_ROWS);
    int*            slot   = (int*)           (ws + OFF_SLOT);
    int*            counts = (int*)           (ws + OFF_COUNTS);
    int*            offs   = (int*)           (ws + OFF_OFFS);
    int*            cursor = (int*)           (ws + OFF_CURS);
    int*            tileM  = (int*)           (ws + OFF_TILEM);
    int*            tileS  = (int*)           (ws + OFF_TILES);
    int*            tcnt   = (int*)           (ws + OFF_TCNT);
    float*          accum  = (float*)         (ws + OFF_ACCUM);

    // zero accumulation targets (graph-capture-safe)
    hipMemsetAsync(energy, 0, (size_t)N_ROWS * M_EXP * sizeof(float), stream);
    hipMemsetAsync(counts, 0, M_EXP * sizeof(int), stream);
    hipMemsetAsync(cursor, 0, M_EXP * sizeof(int), stream);
    hipMemsetAsync(accum,  0, 2 * sizeof(float), stream);
    hipMemsetAsync(xhat,   0, (size_t)N_ROWS * D_DIM * sizeof(float), stream);
    hipMemsetAsync(writes, 0, (size_t)N_ROWS * D_DIM * sizeof(float), stream);

    // 1) normalizations (+ bf16 / transposed copies)
    k_norm_x<<<N_ROWS, 256, 0, stream>>>(x, xn, xnb);
    k_norm_v<<<J_COLS, 256, 0, stream>>>(V, vtb, vnb);
    k_norm_u<<<J_COLS, 256, 0, stream>>>(U, unb, utb);

    // 2) dense WMMA GEMM -> energy
    dim3 ggrid(J_COLS / 64, N_ROWS / 16);
    k_gemm_energy<<<ggrid, 128, 0, stream>>>((const __bf16*)xnb, (const __bf16*)vtb, energy);

    // 3) top-K, per-expert sums, CSR build
    k_topk<<<N_ROWS / 64, 64, 0, stream>>>(energy, topk);
    k_energy_sum<<<M_EXP, 256, 0, stream>>>(energy, esum);
    k_count<<<N_ENT / 256, 256, 0, stream>>>(topk, counts);
    k_scan<<<1, 1, 0, stream>>>(counts, offs, tileM, tileS, tcnt);
    k_fill<<<N_ENT / 256, 256, 0, stream>>>(topk, offs, cursor, rows, slot);

    // 4) expert-blocked WMMA sparse stages
    k_hsparse<<<MAXTILE, 128, 0, stream>>>((const __bf16*)xnb, (const __bf16*)vtb,
                                           tileM, tileS, tcnt, offs, rows, slot, hsd, hsb);
    dim3 sgrid(MAXTILE, D_DIM / 64);
    k_scatter<<<sgrid, 128, 0, stream>>>((const __bf16*)hsb, (const __bf16*)vnb,
                                         (const __bf16*)unb, tileM, tileS, tcnt, offs,
                                         rows, xhat, writes);

    // 5) per-row epilogue (x_out, residual, bf16 writes)
    k_postrow<<<N_ROWS, 256, 0, stream>>>(xn, xhat, writes, wrb, out, accum);

    // 6) h_recon + writer loss (WMMA)
    k_hrecon<<<MAXTILE, 128, 0, stream>>>((const __bf16*)wrb, (const __bf16*)utb,
                                          tileM, tileS, tcnt, offs, rows, slot, hsd, accum);

    // 7) aux scalar
    k_finalize<<<1, 64, 0, stream>>>(esum, accum, out + (size_t)N_ROWS * D_DIM);
}
